// MultiHeadAttention_75909251989864
// MI455X (gfx1250) — compile-verified, matrix operand order checked by reading, not run on hardware
//
#include <hip/hip_runtime.h>
#include <hip/hip_bf16.h>
#include <math.h>

// ---------------------------------------------------------------------------
// MHA forward for MI455X (gfx1250), wave32 + bf16 WMMA 16x16x32.
// Shapes: B=4, N=M=2048, D=1024, H=16, DK=64.
// d_out = [ y : B*N*D floats ][ attn : (H*B)*N*M floats ]
// All WMMA fragments are loaded as contiguous 128-bit chunks:
//   kmap(e,half) = half*8 + e          (e<8)
//                = 16 + half*8 + (e-8) (e>=8)
// so a v16bf fragment = two contiguous v8bf chunks.
// ---------------------------------------------------------------------------

typedef __attribute__((ext_vector_type(16))) __bf16 v16bf;
typedef __attribute__((ext_vector_type(8)))  __bf16 v8bf;
typedef __attribute__((ext_vector_type(8)))  float  v8f;
typedef __attribute__((ext_vector_type(4)))  float  v4f;
typedef int v4i __attribute__((vector_size(16)));   // matches builtin param type

#define B_   4
#define N_   2048
#define M_   2048
#define D_   1024
#define H_   16
#define DK_  64
#define HB_  (H_ * B_)
#define HDK_ (H_ * DK_)     // 1024
#define EPS_ 1e-5f
#define SCALE_ 0.03125f     // 1/sqrt(D_) = 1/32

#define AS1 __attribute__((address_space(1)))
#define AS3 __attribute__((address_space(3)))

#if defined(__has_builtin)
#if __has_builtin(__builtin_amdgcn_global_load_async_to_lds_b128)
#define HAVE_ASYNC_B128 1
#endif
#endif

__device__ __forceinline__ void wait_async0() {
#if __has_builtin(__builtin_amdgcn_s_wait_asynccnt)
    __builtin_amdgcn_s_wait_asynccnt(0);
#else
    asm volatile("s_wait_asynccnt 0x0" ::: "memory");
#endif
}

__device__ __forceinline__ v8f wmma_bf16(v16bf a, v16bf b, v8f c) {
    return __builtin_amdgcn_wmma_f32_16x16x32_bf16(
        /*neg_a=*/false, a, /*neg_b=*/false, b,
        /*c_mod=*/(short)0, c, /*reuse_a=*/false, /*reuse_b=*/false);
}

__device__ __forceinline__ v16bf combine(v8bf lo, v8bf hi) {
    v16bf r;
#pragma unroll
    for (int i = 0; i < 8; ++i) { r[i] = lo[i]; r[i + 8] = hi[i]; }
    return r;
}

// p -> start of this row/col's 32-wide K chunk (bf16, 16B aligned chunks)
__device__ __forceinline__ v16bf load_frag_bf16(const __bf16* p, int half) {
    v8bf lo = *reinterpret_cast<const v8bf*>(p + half * 8);
    v8bf hi = *reinterpret_cast<const v8bf*>(p + 16 + half * 8);
    return combine(lo, hi);
}

// fp32 source: two 32B vector loads + pack-convert to bf16
__device__ __forceinline__ v16bf load_frag_f32(const float* p, int half) {
    v8f lo = *reinterpret_cast<const v8f*>(p + half * 8);
    v8f hi = *reinterpret_cast<const v8f*>(p + 16 + half * 8);
    v16bf r;
#pragma unroll
    for (int i = 0; i < 8; ++i) { r[i] = (__bf16)lo[i]; r[i + 8] = (__bf16)hi[i]; }
    return r;
}

// ---------------------------------------------------------------------------
// 0) Weight conversion. w_q/w_k/w_v -> bf16 TRANSPOSED [h][k][d] (so proj's
//    B fragments are contiguous in d). w_out -> straight bf16 copy (already
//    contiguous in the GEMM K dim).
// ---------------------------------------------------------------------------
__global__ void cvt_weights(const float* __restrict__ wq, const float* __restrict__ wk,
                            const float* __restrict__ wv, const float* __restrict__ wo,
                            __bf16* __restrict__ wqb, __bf16* __restrict__ wkb,
                            __bf16* __restrict__ wvb, __bf16* __restrict__ wob) {
    int i = blockIdx.x * blockDim.x + threadIdx.x;
    if (i < H_ * D_ * DK_) {
        int h  = i >> 16;           // D_*DK_ = 65536
        int rm = i & 65535;
        int d  = rm >> 6;           // DK_ = 64
        int kk = rm & 63;
        int oi = (h << 16) + (kk << 10) + d;   // [h][kk][d]
        wqb[oi] = (__bf16)wq[i];
        wkb[oi] = (__bf16)wk[i];
        wvb[oi] = (__bf16)wv[i];
        wob[i]  = (__bf16)wo[i];
    }
}

// ---------------------------------------------------------------------------
// 1) Per-head projections. grid=(16, H*B, 3), block=256 (8 waves);
//    wave -> 16(N) x 64(DK) tile. qs/ks stored [hb][n][k] (row-major);
//    vs stored TRANSPOSED [hb][v][m] with packed v8bf stores.
// ---------------------------------------------------------------------------
__global__ void proj_kernel(const float* __restrict__ q, const float* __restrict__ k,
                            const float* __restrict__ v,
                            const __bf16* __restrict__ wqb, const __bf16* __restrict__ wkb,
                            const __bf16* __restrict__ wvb,
                            __bf16* __restrict__ qs, __bf16* __restrict__ ks,
                            __bf16* __restrict__ vs) {
    const int t    = blockIdx.z;
    const int hb   = blockIdx.y;
    const int h    = hb / B_;
    const int b    = hb % B_;
    const int wv_  = threadIdx.x >> 5;
    const int lane = threadIdx.x & 31;
    const int half = lane >> 4;
    const int nc   = lane & 15;
    const int n0   = (blockIdx.x * 8 + wv_) * 16;

    const float*  x = (t == 0) ? q   : (t == 1) ? k   : v;
    const __bf16* w = (t == 0) ? wqb : (t == 1) ? wkb : wvb;

    const float*  xrow  = x + (size_t)(b * N_ + n0 + nc) * D_;   // A row (lane&15)
    const __bf16* wbase = w + (size_t)h * DK_ * D_;              // [kk][d]

    v8f acc[4] = {};
    for (int d0 = 0; d0 < D_; d0 += 32) {
        __builtin_prefetch(xrow + d0 + 128, 0, 1);
        v16bf a = load_frag_f32(xrow + d0, half);
#pragma unroll
        for (int j = 0; j < 4; ++j) {
            v16bf bb = load_frag_bf16(wbase + (size_t)(j * 16 + nc) * D_ + d0, half);
            acc[j] = wmma_bf16(a, bb, acc[j]);
        }
    }

    if (t == 2) {
        // vs transposed [hb][v][m]: C fragment rows are contiguous -> v8bf stores
        __bf16* obase = vs + (size_t)hb * DK_ * M_;
#pragma unroll
        for (int j = 0; j < 4; ++j) {
            v8bf pk;
#pragma unroll
            for (int r8 = 0; r8 < 8; ++r8) pk[r8] = (__bf16)acc[j][r8];
            *reinterpret_cast<v8bf*>(obase + (size_t)(j * 16 + nc) * M_ + n0 + half * 8) = pk;
        }
    } else {
        __bf16* orow = ((t == 0) ? qs : ks) + (size_t)(hb * N_ + n0) * DK_;
#pragma unroll
        for (int j = 0; j < 4; ++j)
#pragma unroll
            for (int r8 = 0; r8 < 8; ++r8)
                orow[(size_t)(r8 + half * 8) * DK_ + j * 16 + nc] = (__bf16)acc[j][r8];
    }
}

// ---------------------------------------------------------------------------
// 2) Scores: attn[hb,n,m] = SCALE * qs[hb,n,:] . ks[hb,m,:]  (K=64 -> 2 WMMAs)
//    grid=(N/16, H*B), block=256. The 16x64 q-tile (2KB) is staged to LDS
//    with async-to-LDS b128, fragments built from LDS (ds_load_b128).
// ---------------------------------------------------------------------------
__global__ void scores_kernel(const __bf16* __restrict__ qs, const __bf16* __restrict__ ks,
                              float* __restrict__ attn) {
    const int hb   = blockIdx.y;
    const int tid  = threadIdx.x;
    const int wv_  = tid >> 5;
    const int lane = tid & 31;
    const int half = lane >> 4;
    const int nc   = lane & 15;
    const int n0   = blockIdx.x * 16;

    __shared__ __bf16 qtile[16 * DK_];   // 2KB, shared by all 8 waves
    const __bf16* qrow = qs + (size_t)(hb * N_ + n0) * DK_;

#ifdef HAVE_ASYNC_B128
    if (tid < 128)
        __builtin_amdgcn_global_load_async_to_lds_b128(
            (AS1 v4i*)(qrow + tid * 8), (AS3 v4i*)(qtile + tid * 8), 0, 0);
    wait_async0();
    __syncthreads();
#else
    if (tid < 128)
        *reinterpret_cast<v8bf*>(qtile + tid * 8) =
            *reinterpret_cast<const v8bf*>(qrow + tid * 8);
    __syncthreads();
#endif

    const v16bf a0 = load_frag_bf16(qtile + nc * DK_, half);
    const v16bf a1 = load_frag_bf16(qtile + nc * DK_ + 32, half);

    const __bf16* kbase = ks + (size_t)hb * M_ * DK_;
    for (int mt = wv_; mt < M_ / 64; mt += 8) {
        const int m0 = mt * 64;
#pragma unroll
        for (int j = 0; j < 4; ++j) {
            const __bf16* kr = kbase + (size_t)(m0 + j * 16 + nc) * DK_;
            v16bf b0 = load_frag_bf16(kr, half);
            v16bf b1 = load_frag_bf16(kr + 32, half);
            v8f acc = {};
            acc = wmma_bf16(a0, b0, acc);
            acc = wmma_bf16(a1, b1, acc);
#pragma unroll
            for (int r8 = 0; r8 < 8; ++r8)
                attn[(size_t)(hb * N_ + n0 + r8 + half * 8) * M_ + m0 + j * 16 + nc] =
                    acc[r8] * SCALE_;
        }
    }
}

// ---------------------------------------------------------------------------
// 3) Mask + row softmax in place; vectorized (float4 x2 + 8-byte mask reads).
//    grid = H*B*N blocks, 256 threads/row (each thread owns 8 contiguous).
// ---------------------------------------------------------------------------
__global__ void softmax_kernel(float* __restrict__ attn, const unsigned char* __restrict__ mask) {
    const int r  = blockIdx.x;           // 0 .. HB*N-1
    const int hb = r / N_;
    const int n  = r - hb * N_;
    const int b  = hb % B_;              // hb = h*B + b
    float* row = attn + (size_t)r * M_;
    const unsigned char* mrow = mask + (size_t)(b * N_ + n) * M_;
    const int tid = threadIdx.x;

    v4f x0 = *reinterpret_cast<const v4f*>(row + tid * 8);
    v4f x1 = *reinterpret_cast<const v4f*>(row + tid * 8 + 4);
    const unsigned long long mb =
        *reinterpret_cast<const unsigned long long*>(mrow + tid * 8);

    float xv[8];
#pragma unroll
    for (int i = 0; i < 4; ++i) { xv[i] = x0[i]; xv[i + 4] = x1[i]; }
    float lmax = -INFINITY;
#pragma unroll
    for (int i = 0; i < 8; ++i) {
        if ((mb >> (8 * i)) & 0xffull) xv[i] = -INFINITY;
        lmax = fmaxf(lmax, xv[i]);
    }

    __shared__ float red[256];
    red[tid] = lmax;
    __syncthreads();
    for (int s = 128; s > 0; s >>= 1) {
        if (tid < s) red[tid] = fmaxf(red[tid], red[tid + s]);
        __syncthreads();
    }
    const float gmax = red[0];
    __syncthreads();

    float lsum = 0.f;
#pragma unroll
    for (int i = 0; i < 8; ++i) { xv[i] = __expf(xv[i] - gmax); lsum += xv[i]; }
    red[tid] = lsum;
    __syncthreads();
    for (int s = 128; s > 0; s >>= 1) {
        if (tid < s) red[tid] += red[tid + s];
        __syncthreads();
    }
    const float inv = 1.f / red[0];

#pragma unroll
    for (int i = 0; i < 4; ++i) { x0[i] = xv[i] * inv; x1[i] = xv[i + 4] * inv; }
    *reinterpret_cast<v4f*>(row + tid * 8)     = x0;
    *reinterpret_cast<v4f*>(row + tid * 8 + 4) = x1;
}

// ---------------------------------------------------------------------------
// 4) ctx = P @ V (K = M = 2048 -> 64 steps). P: fp32 rows (vector load+cvt),
//    V: transposed [hb][v][m] -> contiguous B fragments. ctx row-major
//    [b*N+n][h*64+v]. grid=(16, H*B), block 256.
// ---------------------------------------------------------------------------
__global__ void pv_kernel(const float* __restrict__ attn, const __bf16* __restrict__ vs,
                          __bf16* __restrict__ ctx) {
    const int hb   = blockIdx.y;
    const int h    = hb / B_;
    const int b    = hb % B_;
    const int wv_  = threadIdx.x >> 5;
    const int lane = threadIdx.x & 31;
    const int half = lane >> 4;
    const int nc   = lane & 15;
    const int n0   = (blockIdx.x * 8 + wv_) * 16;

    const float*  prow  = attn + (size_t)(hb * N_ + n0 + nc) * M_;
    const __bf16* vbase = vs + (size_t)hb * DK_ * M_;    // [v][m]

    v8f acc[4] = {};
    for (int m0 = 0; m0 < M_; m0 += 32) {
        __builtin_prefetch(prow + m0 + 128, 0, 1);
        v16bf a = load_frag_f32(prow + m0, half);
#pragma unroll
        for (int j = 0; j < 4; ++j) {
            v16bf bb = load_frag_bf16(vbase + (size_t)(j * 16 + nc) * M_ + m0, half);
            acc[j] = wmma_bf16(a, bb, acc[j]);
        }
    }
#pragma unroll
    for (int j = 0; j < 4; ++j)
#pragma unroll
        for (int r8 = 0; r8 < 8; ++r8)
            ctx[(size_t)(b * N_ + n0 + r8 + half * 8) * HDK_ + h * DK_ + j * 16 + nc] =
                (__bf16)acc[j][r8];
}

// ---------------------------------------------------------------------------
// 5) x = ctx @ w_out^T + q  ([B*N x 1024] x [1024 x 1024]). Both A (ctx rows)
//    and B (w_out rows) are contiguous in K -> pure b128 fragment loads.
//    grid = (64, 16).
// ---------------------------------------------------------------------------
__global__ void outproj_kernel(const __bf16* __restrict__ ctx, const __bf16* __restrict__ wob,
                               const float* __restrict__ q, float* __restrict__ xout) {
    const int wv_  = threadIdx.x >> 5;
    const int lane = threadIdx.x & 31;
    const int half = lane >> 4;
    const int nc   = lane & 15;
    const int row0 = (blockIdx.x * 8 + wv_) * 16;   // over B*N = 8192 rows
    const int col0 = blockIdx.y * 64;

    const __bf16* arow = ctx + (size_t)(row0 + nc) * HDK_;

    v8f acc[4] = {};
    for (int k0 = 0; k0 < HDK_; k0 += 32) {
        v16bf a = load_frag_bf16(arow + k0, half);
#pragma unroll
        for (int j = 0; j < 4; ++j) {
            v16bf bb = load_frag_bf16(wob + (size_t)(col0 + j * 16 + nc) * HDK_ + k0, half);
            acc[j] = wmma_bf16(a, bb, acc[j]);
        }
    }
#pragma unroll
    for (int j = 0; j < 4; ++j)
#pragma unroll
        for (int r8 = 0; r8 < 8; ++r8) {
            size_t idx = (size_t)(row0 + r8 + half * 8) * D_ + col0 + j * 16 + nc;
            xout[idx] = acc[j][r8] + q[idx];   // residual
        }
}

// ---------------------------------------------------------------------------
// 6) In-place LayerNorm over D per row (vectorized float4).
//    grid = B*N blocks, 256 threads (each owns 4 contiguous elements).
// ---------------------------------------------------------------------------
__global__ void ln_kernel(float* __restrict__ y, const float* __restrict__ gamma,
                          const float* __restrict__ beta) {
    const int r = blockIdx.x;
    float* row = y + (size_t)r * D_;
    const int tid = threadIdx.x;

    v4f xv = *reinterpret_cast<const v4f*>(row + tid * 4);
    float s = xv[0] + xv[1] + xv[2] + xv[3];

    __shared__ float red[256];
    red[tid] = s;
    __syncthreads();
    for (int st = 128; st > 0; st >>= 1) {
        if (tid < st) red[tid] += red[tid + st];
        __syncthreads();
    }
    const float mu = red[0] * (1.0f / D_);
    __syncthreads();

    float vsum = 0.f;
#pragma unroll
    for (int i = 0; i < 4; ++i) { float d = xv[i] - mu; vsum += d * d; }
    red[tid] = vsum;
    __syncthreads();
    for (int st = 128; st > 0; st >>= 1) {
        if (tid < st) red[tid] += red[tid + st];
        __syncthreads();
    }
    const float rstd = rsqrtf(red[0] * (1.0f / D_) + EPS_);

    const v4f g = *reinterpret_cast<const v4f*>(gamma + tid * 4);
    const v4f be = *reinterpret_cast<const v4f*>(beta + tid * 4);
    v4f o;
#pragma unroll
    for (int i = 0; i < 4; ++i) o[i] = (xv[i] - mu) * rstd * g[i] + be[i];
    *reinterpret_cast<v4f*>(row + tid * 4) = o;
}

// ---------------------------------------------------------------------------
extern "C" void kernel_launch(void* const* d_in, const int* in_sizes, int n_in,
                              void* d_out, int out_size, void* d_ws, size_t ws_size,
                              hipStream_t stream) {
    const float* q     = (const float*)d_in[0];
    const float* k     = (const float*)d_in[1];
    const float* v     = (const float*)d_in[2];
    const unsigned char* mask = (const unsigned char*)d_in[3];
    const float* w_q   = (const float*)d_in[4];
    const float* w_k   = (const float*)d_in[5];
    const float* w_v   = (const float*)d_in[6];
    const float* w_out = (const float*)d_in[7];
    const float* gamma = (const float*)d_in[8];
    const float* beta  = (const float*)d_in[9];

    // Workspace (bf16): 4 weights (2MB each) + qs/ks/vs (16MB each) + ctx (16MB)
    char* ws = (char*)d_ws;
    const size_t WSZ = (size_t)H_ * D_ * DK_;        // 1,048,576 elems
    const size_t PSZ = (size_t)HB_ * N_ * DK_;       // 8,388,608 elems
    __bf16* wqb = (__bf16*)ws;  ws += WSZ * sizeof(__bf16);
    __bf16* wkb = (__bf16*)ws;  ws += WSZ * sizeof(__bf16);
    __bf16* wvb = (__bf16*)ws;  ws += WSZ * sizeof(__bf16);
    __bf16* wob = (__bf16*)ws;  ws += WSZ * sizeof(__bf16);
    __bf16* qs  = (__bf16*)ws;  ws += PSZ * sizeof(__bf16);
    __bf16* ks  = (__bf16*)ws;  ws += PSZ * sizeof(__bf16);
    __bf16* vs  = (__bf16*)ws;  ws += PSZ * sizeof(__bf16);   // transposed [hb][v][m]
    __bf16* ctx = (__bf16*)ws;  ws += (size_t)B_ * N_ * HDK_ * sizeof(__bf16);

    float* y    = (float*)d_out;                       // [B,N,D]
    float* attn = y + (size_t)B_ * N_ * D_;            // [H*B,N,M]

    cvt_weights<<<(int)(WSZ / 256), 256, 0, stream>>>(w_q, w_k, w_v, w_out, wqb, wkb, wvb, wob);
    proj_kernel<<<dim3(16, HB_, 3), 256, 0, stream>>>(q, k, v, wqb, wkb, wvb, qs, ks, vs);
    scores_kernel<<<dim3(N_ / 16, HB_), 256, 0, stream>>>(qs, ks, attn);
    softmax_kernel<<<HB_ * N_, 256, 0, stream>>>(attn, mask);
    pv_kernel<<<dim3(16, HB_), 256, 0, stream>>>(attn, vs, ctx);
    outproj_kernel<<<dim3(64, 16), 256, 0, stream>>>(ctx, wob, q, y);
    ln_kernel<<<B_ * N_, 256, 0, stream>>>(y, gamma, beta);
}
